// LinformerMultiHeadAttention_10857677324404
// MI455X (gfx1250) — compile-verified, matrix operand order checked by reading, not run on hardware
//
#include <hip/hip_runtime.h>
#include <math.h>

#define DEV __device__ __forceinline__

typedef __bf16 v8bf  __attribute__((ext_vector_type(8)));
typedef __bf16 v16bf __attribute__((ext_vector_type(16)));
typedef float  v8f   __attribute__((ext_vector_type(8)));
typedef int    v4i   __attribute__((vector_size(4 * sizeof(int))));

union Frag { v16bf f; v8bf h[2]; };

static constexpr int cH = 16;
static constexpr int cD = 64;
static constexpr int cC = 1024;
static constexpr int cK = 256;
static constexpr int cN = 4096;
static constexpr int cB = 2;

#if __has_builtin(__builtin_amdgcn_global_load_async_to_lds_b128) && \
    __has_builtin(__builtin_amdgcn_s_wait_asynccnt)
#define USE_ASYNC_LDS 1
typedef __attribute__((address_space(1))) v4i* gptr_v4i;
typedef __attribute__((address_space(3))) v4i* lptr_v4i;
#endif

DEV __bf16 f2bf(float x) {
  unsigned u = __builtin_bit_cast(unsigned, x);
  unsigned r = u + 0x7FFFu + ((u >> 16) & 1u);   // round-to-nearest-even
  return __builtin_bit_cast(__bf16, (unsigned short)(r >> 16));
}

DEV float geluf(float x) {
  return 0.5f * x * (1.0f + erff(x * 0.70710678118654752f));
}

DEV v8f zero8() { v8f z = {0.f,0.f,0.f,0.f,0.f,0.f,0.f,0.f}; return z; }

// A-frag and B-frag for v_wmma_f32_16x16x32_bf16 share the per-lane pattern
// when both matrices are row-major [16 x K] with contiguous K:
//   lane m (0-15)  : row m, K = {0..7, 16..23}
//   lane m+16      : row m, K = {8..15, 24..31}
// Pointers below are pre-offset by (row*ld + half*8).
DEV Frag load_frag_at(const __bf16* p) {
  Frag fr;
  fr.h[0] = *(const v8bf*)(p);
  fr.h[1] = *(const v8bf*)(p + 16);
  return fr;
}

// ---------------------------------------------------------------- convert
__global__ void cvt_f32_bf16_kernel(const float* __restrict__ s,
                                    __bf16* __restrict__ d, long n) {
  long i = (long)blockIdx.x * blockDim.x + threadIdx.x;
  long stride = (long)gridDim.x * blockDim.x;
  for (; i < n; i += stride) d[i] = f2bf(s[i]);
}

// ---------------------------------------------------------------- generic GEMM
// C = A * B^T ; A:[M,Kr] row-major, B:[Nc,Kr] row-major, both bf16.
// Register-blocked: one wave computes a 32x64 output block = 2x4 WMMA tiles:
// per K-step, 6 fragment loads feed 8 v_wmma (24B/lane/WMMA from cache).
// __launch_bounds__(256,2): relax the occupancy-driven VGPR cap so the
// 64 accumulator + 48 fragment VGPRs live without scratch spills.
// Requires M % 32 == 0, Nc % 64 == 0 (true for all pipeline shapes).
// aMode/bMode: batch index from blockIdx.z: 0 -> z, 1 -> z/H, 2 -> z%H.
// Output element (m,n) written at cbase + m*cLdRow + n*cLdCol (allows transpose).
// cSplit: cbase = (z/H)*cStride + (z%H)*cStride2, else z*cStride.
// bias (f32) indexed by (z%H)*biasStrideH + n. act: 0 none, 1 gelu.
// outF: 0 -> bf16 store to Cb, 1 -> f32 store to Cf.
__global__ void __launch_bounds__(256, 2)
gemm_bt_kernel(const __bf16* __restrict__ A,
               const __bf16* __restrict__ B,
               __bf16* __restrict__ Cb,
               float* __restrict__ Cf,
               int M, int Nc, int Kr,
               long aStride, int aMode,
               long bStride, int bMode,
               long cStride, long cStride2, int cSplit,
               int cLdRow, int cLdCol,
               const float* __restrict__ bias, int biasStrideH,
               int act, int outF) {
  const int z    = blockIdx.z;
  const int tid  = threadIdx.x;
  const int lane = tid & 31;
  const int r    = lane & 15;
  const int half = lane >> 4;

  const int superN = Nc >> 6;     // 64-col blocks
  const int superM = M >> 5;      // 32-row blocks
  const int waveId = blockIdx.x * 8 + (tid >> 5);
  if (waveId >= superM * superN) return;   // wave-uniform guard (EXEC all-1s for WMMA)
  const int sm = waveId / superN;
  const int sn = waveId % superN;

  const long ai = (aMode == 0) ? z : ((aMode == 1) ? (z / cH) : (z % cH));
  const long bi = (bMode == 0) ? z : ((bMode == 1) ? (z / cH) : (z % cH));
  const __bf16* Ab = A + ai * aStride;
  const __bf16* Bb = B + bi * bStride;

  const __bf16* Ar[2];
#pragma unroll
  for (int im = 0; im < 2; ++im)
    Ar[im] = Ab + (long)(sm * 32 + im * 16 + r) * Kr + half * 8;
  const __bf16* Br[4];
#pragma unroll
  for (int in = 0; in < 4; ++in)
    Br[in] = Bb + (long)(sn * 64 + in * 16 + r) * Kr + half * 8;

  v8f acc[2][4];
#pragma unroll
  for (int im = 0; im < 2; ++im)
#pragma unroll
    for (int in = 0; in < 4; ++in) acc[im][in] = zero8();

  for (int k = 0; k < Kr; k += 32) {
    if (k + 32 < Kr) {                        // pull next K-slab toward the WGP
      __builtin_prefetch(Ar[0] + k + 32, 0, 3);
      __builtin_prefetch(Ar[1] + k + 32, 0, 3);
      __builtin_prefetch(Br[0] + k + 32, 0, 3);
      __builtin_prefetch(Br[2] + k + 32, 0, 3);
    }
    Frag a[2], b[4];
#pragma unroll
    for (int im = 0; im < 2; ++im) a[im] = load_frag_at(Ar[im] + k);
#pragma unroll
    for (int in = 0; in < 4; ++in) b[in] = load_frag_at(Br[in] + k);
#pragma unroll
    for (int im = 0; im < 2; ++im)
#pragma unroll
      for (int in = 0; in < 4; ++in)
        acc[im][in] = __builtin_amdgcn_wmma_f32_16x16x32_bf16(
            false, a[im].f, false, b[in].f, (short)0, acc[im][in], false, false);
  }

  const long cb = cSplit ? ((long)(z / cH) * cStride + (long)(z % cH) * cStride2)
                         : (long)z * cStride;
#pragma unroll
  for (int in = 0; in < 4; ++in) {
    const int n = sn * 64 + in * 16 + r;
    const float bv = bias ? bias[(long)(z % cH) * biasStrideH + n] : 0.0f;
#pragma unroll
    for (int im = 0; im < 2; ++im) {
#pragma unroll
      for (int i = 0; i < 8; ++i) {
        const int m = sm * 32 + im * 16 + half * 8 + i;
        float v = acc[im][in][i] + bv;
        if (act == 1) v = geluf(v);
        const long off = cb + (long)m * cLdRow + (long)n * cLdCol;
        if (outF) Cf[off] = v;
        else      Cb[off] = f2bf(v);
      }
    }
  }
}

// ---------------------------------------------------------------- fused scores+softmax
// scores = (Q[b,h] (N x D)) * (Ket[b,h] (K x D))^T * (1/sqrt(D)); softmax over K; P bf16.
// Block: 128 threads = 4 waves; block owns a 16-row strip x all 256 cols.
// Ket[z] (256x64 bf16 = 32KB) is staged once per block into LDS via the CDNA5
// async-copy engine (GLOBAL_LOAD_ASYNC_TO_LDS_B128, ASYNCcnt), then all WMMA
// B-fragments are served by ds_load_b128 out of the 320KB/WGP LDS.
__global__ void __launch_bounds__(128, 2)
scores_softmax_kernel(const __bf16* __restrict__ Q,
                      const __bf16* __restrict__ Ket,
                      __bf16* __restrict__ P) {
  const int z       = blockIdx.z;          // b*H + h
  const int rowBase = blockIdx.x * 16;
  const int tid     = threadIdx.x;
  const int w       = tid >> 5;
  const int lane    = tid & 31;
  const int r       = lane & 15;
  const int half    = lane >> 4;

  __shared__ __bf16 ketLds[cK * cD];       // 32KB
  __shared__ float  wmax[64];
  __shared__ float  wsum[64];

  const __bf16* Qb = Q   + (long)z * cN * cD;
  const __bf16* Kg = Ket + (long)z * cK * cD;

  // ---- stage Ket[z] into LDS: 2048 x 16B chunks, 128 threads, 16 iters
#pragma unroll
  for (int it = 0; it < 16; ++it) {
    const int idx = it * 128 + tid;
    const __bf16* g = Kg + (long)idx * 8;
    __bf16* l = &ketLds[idx * 8];
#ifdef USE_ASYNC_LDS
    __builtin_amdgcn_global_load_async_to_lds_b128(
        (gptr_v4i)(__bf16*)g, (lptr_v4i)l, 0, 0);
#else
    *(v8bf*)l = *(const v8bf*)g;
#endif
  }
#ifdef USE_ASYNC_LDS
  __builtin_amdgcn_s_wait_asynccnt(0);
#endif
  __syncthreads();

  v8f acc[4];
#pragma unroll
  for (int t = 0; t < 4; ++t) acc[t] = zero8();

#pragma unroll
  for (int kd = 0; kd < cD; kd += 32) {
    Frag a = load_frag_at(Qb + (long)(rowBase + r) * cD + kd + half * 8);
#pragma unroll
    for (int t = 0; t < 4; ++t) {
      const int col = w * 64 + t * 16;
      Frag b = load_frag_at(&ketLds[(col + r) * cD + kd + half * 8]);
      acc[t] = __builtin_amdgcn_wmma_f32_16x16x32_bf16(false, a.f, false, b.f,
                                                       (short)0, acc[t], false, false);
    }
  }

  const float sc = 0.125f;                  // 1/sqrt(64)
#pragma unroll
  for (int t = 0; t < 4; ++t)
#pragma unroll
    for (int i = 0; i < 8; ++i) acc[t][i] *= sc;

  // per-wave row max: lane holds (row = i + 8*half) for col = r + tile base
  float vmax[8];
#pragma unroll
  for (int i = 0; i < 8; ++i)
    vmax[i] = fmaxf(fmaxf(acc[0][i], acc[1][i]), fmaxf(acc[2][i], acc[3][i]));
#pragma unroll
  for (int off = 1; off < 16; off <<= 1)
#pragma unroll
    for (int i = 0; i < 8; ++i)
      vmax[i] = fmaxf(vmax[i], __shfl_xor(vmax[i], off, 32));
  if (r == 0) {
#pragma unroll
    for (int i = 0; i < 8; ++i) wmax[w * 16 + half * 8 + i] = vmax[i];
  }
  __syncthreads();

  float rmax[8];
#pragma unroll
  for (int i = 0; i < 8; ++i) {
    const int row = half * 8 + i;
    rmax[i] = fmaxf(fmaxf(wmax[row], wmax[16 + row]),
                    fmaxf(wmax[32 + row], wmax[48 + row]));
  }

  float vsum[8];
#pragma unroll
  for (int i = 0; i < 8; ++i) vsum[i] = 0.0f;
#pragma unroll
  for (int t = 0; t < 4; ++t)
#pragma unroll
    for (int i = 0; i < 8; ++i) {
      const float e = __expf(acc[t][i] - rmax[i]);
      acc[t][i] = e;
      vsum[i] += e;
    }
#pragma unroll
  for (int off = 1; off < 16; off <<= 1)
#pragma unroll
    for (int i = 0; i < 8; ++i) vsum[i] += __shfl_xor(vsum[i], off, 32);
  if (r == 0) {
#pragma unroll
    for (int i = 0; i < 8; ++i) wsum[w * 16 + half * 8 + i] = vsum[i];
  }
  __syncthreads();

  float rinv[8];
#pragma unroll
  for (int i = 0; i < 8; ++i) {
    const int row = half * 8 + i;
    const float s = wsum[row] + wsum[16 + row] + wsum[32 + row] + wsum[48 + row];
    rinv[i] = __frcp_rn(s);
  }

  __bf16* Pz = P + (long)z * cN * cK;
#pragma unroll
  for (int t = 0; t < 4; ++t) {
    const int col = w * 64 + t * 16 + r;
#pragma unroll
    for (int i = 0; i < 8; ++i) {
      const int m = rowBase + half * 8 + i;
      Pz[(long)m * cK + col] = f2bf(acc[t][i] * rinv[i]);
    }
  }
}

// ---------------------------------------------------------------- host
extern "C" void kernel_launch(void* const* d_in, const int* in_sizes, int n_in,
                              void* d_out, int out_size, void* d_ws, size_t ws_size,
                              hipStream_t stream) {
  const float* x  = (const float*)d_in[0];
  const float* Wq = (const float*)d_in[1];
  const float* Wk = (const float*)d_in[2];
  const float* Wv = (const float*)d_in[3];
  const float* We = (const float*)d_in[4];
  const float* bE = (const float*)d_in[5];
  const float* Wf = (const float*)d_in[6];
  const float* bF = (const float*)d_in[7];
  const float* Wo = (const float*)d_in[8];
  const float* bo = (const float*)d_in[9];
  float* out = (float*)d_out;

  const long nx  = (long)cB * cN * cC;       // 8.39M
  const long nWp = (long)cH * cD * cC;       // 1.05M  (Wq/Wk/Wv, Wo)
  const long nWe = (long)cH * cK * cN;       // 16.8M  (We/Wf)
  const long nQ  = (long)cB * cH * cN * cD;  // 8.39M  (Q/Kt/Vt, G)
  const long nKe = (long)cB * cH * cK * cD;  // 0.52M  (Ket/Vf)
  const long nP  = (long)cB * cH * cN * cK;  // 33.6M

  char* p = (char*)d_ws;
  auto bump = [&](long elems) { __bf16* q = (__bf16*)p;
                                p += ((elems * 2 + 255) & ~255L); return q; };
  __bf16* xb  = bump(nx);
  __bf16* Wqb = bump(nWp);
  __bf16* Wkb = bump(nWp);
  __bf16* Wvb = bump(nWp);
  __bf16* Web = bump(nWe);
  __bf16* Wfb = bump(nWe);
  __bf16* Wob = bump(nWp);
  __bf16* Qb  = bump(nQ);
  __bf16* Ktb = bump(nQ);
  __bf16* Vtb = bump(nQ);
  __bf16* Ketb= bump(nKe);
  __bf16* Vfb = bump(nKe);
  __bf16* Pb  = bump(nP);
  __bf16* Gb  = bump(nQ);

  // 1) f32 -> bf16 conversions
  cvt_f32_bf16_kernel<<<2048, 256, 0, stream>>>(x,  xb,  nx);
  cvt_f32_bf16_kernel<<<512,  256, 0, stream>>>(Wq, Wqb, nWp);
  cvt_f32_bf16_kernel<<<512,  256, 0, stream>>>(Wk, Wkb, nWp);
  cvt_f32_bf16_kernel<<<512,  256, 0, stream>>>(Wv, Wvb, nWp);
  cvt_f32_bf16_kernel<<<2048, 256, 0, stream>>>(We, Web, nWe);
  cvt_f32_bf16_kernel<<<2048, 256, 0, stream>>>(Wf, Wfb, nWe);
  cvt_f32_bf16_kernel<<<512,  256, 0, stream>>>(Wo, Wob, nWp);

  auto gemm = [&](const __bf16* A, const __bf16* B, __bf16* Cb, float* Cf,
                  int M, int Nc, int Kr, long aS, int aM, long bS, int bM,
                  long cS, long cS2, int cSplit, int cLdR, int cLdC,
                  const float* bias, int biasSH, int act, int outF, int Z) {
    const int tiles  = (M / 32) * (Nc / 64);
    const int blocks = (tiles + 7) / 8;
    gemm_bt_kernel<<<dim3(blocks, 1, Z), 256, 0, stream>>>(
        A, B, Cb, Cf, M, Nc, Kr, aS, aM, bS, bM, cS, cS2, cSplit,
        cLdR, cLdC, bias, biasSH, act, outF);
  };

  const int Z = cB * cH;

  // 2) Q = x @ Wq^T  -> Q[b,h,n,d]
  gemm(xb, Wqb, Qb, nullptr, cN, cD, cC, (long)cN * cC, 1, (long)cD * cC, 2,
       (long)cN * cD, 0, 0, cD, 1, nullptr, 0, 0, 0, Z);
  // 3) Kt[b,h,d,n] (transposed write)
  gemm(xb, Wkb, Ktb, nullptr, cN, cD, cC, (long)cN * cC, 1, (long)cD * cC, 2,
       (long)cD * cN, 0, 0, 1, cN, nullptr, 0, 0, 0, Z);
  // 4) Vt[b,h,d,n]
  gemm(xb, Wvb, Vtb, nullptr, cN, cD, cC, (long)cN * cC, 1, (long)cD * cC, 2,
       (long)cD * cN, 0, 0, 1, cN, nullptr, 0, 0, 0, Z);
  // 5) Ket[b,h,k,d] = (Kt @ We^T)^T + bE   (compute (d,k), write transposed, bias by n=k)
  gemm(Ktb, Web, Ketb, nullptr, cD, cK, cN, (long)cD * cN, 0, (long)cK * cN, 2,
       (long)cK * cD, 0, 0, 1, cD, bE, cK, 0, 0, Z);
  // 6) Vf[b,h,d,k] = Vt @ Wf^T + bF
  gemm(Vtb, Wfb, Vfb, nullptr, cD, cK, cN, (long)cD * cN, 0, (long)cK * cN, 2,
       (long)cD * cK, 0, 0, cK, 1, bF, cK, 0, 0, Z);

  // 7) fused scores * 1/sqrt(D) -> softmax over K -> P[b,h,n,k] (bf16)
  scores_softmax_kernel<<<dim3(cN / 16, 1, Z), 128, 0, stream>>>(Qb, Ketb, Pb);

  // 8) out = P @ Vf^T, gelu, head-concat -> G[b,n,h*64+d]
  gemm(Pb, Vfb, Gb, nullptr, cN, cD, cK, (long)cN * cK, 0, (long)cD * cK, 0,
       (long)cN * cH * cD, (long)cD, 1, cH * cD, 1, nullptr, 0, 1, 0, Z);

  // 9) y = G @ Wo^T + bo  (f32 output)
  gemm(Gb, Wob, nullptr, out, cB * cN, cC, cH * cD, 0, 0, 0, 0,
       0, 0, 0, cC, 1, bo, 0, 0, 1, 1);
}
